// Decoder_87943750353304
// MI455X (gfx1250) — matrix operand order repaired and checked by reading, not compiled, not run on hardware
//
#include <hip/hip_runtime.h>

// ---------------------------------------------------------------------------
// CDNA5 (gfx1250) attention-LSTM decoder.
// B=64, T=512, L=128, EPROJS=DUNITS=ATT_DIM=1024, EMB_DIM=512, NGATES=4096.
// bf16 WMMA (v_wmma_f32_16x16x32_bf16) for all GEMMs, fp32 accumulation.
// B operands pre-packed into WMMA fragment order -> all fragment loads are
// global_load_b128; each wave computes a 64x16 strip (4 WMMA per B fragment).
// ---------------------------------------------------------------------------

typedef unsigned short u16;
typedef __attribute__((ext_vector_type(16))) __bf16 v16bf;
typedef __attribute__((ext_vector_type(8)))  __bf16 v8bf;
typedef __attribute__((ext_vector_type(8)))  float  v8f;

#define B_ 64
#define T_ 512
#define L_ 128
#define DU_ 1024
#define EMB_ 512
#define NG_ 4096

__device__ __forceinline__ u16 f2bf(float x) {
  unsigned u = __float_as_uint(x);
  u += 0x7FFFu + ((u >> 16) & 1u);          // round-to-nearest-even
  return (u16)(u >> 16);
}
__device__ __forceinline__ float bf2f(u16 v) {
  return __uint_as_float(((unsigned)v) << 16);
}
__device__ __forceinline__ float sigm(float x) {
  return 1.0f / (1.0f + __expf(-x));
}

// ---------------- WMMA fragment loaders ------------------------------------
// A 16x32 (MxK) from row-major: lane holds K run {kb..kb+15}, kb = k0 + 16*(lane/16)
// -> two contiguous 16B loads (global_load_b128).
__device__ __forceinline__ v16bf load_frag_a(const u16* A, long lda,
                                             int m, int k0, int lane) {
  const u16* p = A + (long)(m + (lane & 15)) * lda + k0 + ((lane & 16) ? 16 : 0);
  v8bf lo = *reinterpret_cast<const v8bf*>(p);
  v8bf hi = *reinterpret_cast<const v8bf*>(p + 8);
  return __builtin_shufflevector(lo, hi, 0, 1, 2, 3, 4, 5, 6, 7,
                                 8, 9, 10, 11, 12, 13, 14, 15);
}
// B from packed fragment layout: tile (k0/32, n0/16) stored as 32 lanes x 16
// bf16 contiguous (1KB). Lane reads 32B contiguous -> two b128 loads.
__device__ __forceinline__ v16bf load_frag_bp(const u16* p) {
  v8bf lo = *reinterpret_cast<const v8bf*>(p);
  v8bf hi = *reinterpret_cast<const v8bf*>(p + 8);
  return __builtin_shufflevector(lo, hi, 0, 1, 2, 3, 4, 5, 6, 7,
                                 8, 9, 10, 11, 12, 13, 14, 15);
}

__device__ __forceinline__ void accum4(const u16* __restrict__ A, long lda,
                                       const u16* __restrict__ P, int K,
                                       int ntN, int m0, int n0, int lane,
                                       v8f& c0, v8f& c1, v8f& c2, v8f& c3) {
  const long tstride = (long)ntN << 9;  // packed elems per K-step (ntN tiles * 512)
  const u16* bp = P + ((long)(n0 >> 4) << 9) + lane * 16;
  for (int k0 = 0; k0 < K; k0 += 32) {
    v16bf b = load_frag_bp(bp);
    __builtin_prefetch(bp + tstride, 0, 1);  // next K-step tile (linear walk)
    v16bf a0 = load_frag_a(A, lda, m0, k0, lane);
    c0 = __builtin_amdgcn_wmma_f32_16x16x32_bf16(false, a0, false, b, (short)0, c0, false, false);
    v16bf a1 = load_frag_a(A, lda, m0 + 16, k0, lane);
    c1 = __builtin_amdgcn_wmma_f32_16x16x32_bf16(false, a1, false, b, (short)0, c1, false, false);
    v16bf a2 = load_frag_a(A, lda, m0 + 32, k0, lane);
    c2 = __builtin_amdgcn_wmma_f32_16x16x32_bf16(false, a2, false, b, (short)0, c2, false, false);
    v16bf a3 = load_frag_a(A, lda, m0 + 48, k0, lane);
    c3 = __builtin_amdgcn_wmma_f32_16x16x32_bf16(false, a3, false, b, (short)0, c3, false, false);
    bp += tstride;
  }
}

// C = A0@B0 + A1@B1 + A2@B2 + bias ; one wave per 64x16 strip, 4 waves/block.
// B matrices are pre-packed fragment tiles keyed by (K/32, N/16).
__global__ void wmma_gemm3(const u16* __restrict__ A0, long lda0,
                           const u16* __restrict__ B0, int K0,
                           const u16* __restrict__ A1, long lda1,
                           const u16* __restrict__ B1, int K1,
                           const u16* __restrict__ A2, long lda2,
                           const u16* __restrict__ B2, int K2,
                           const float* __restrict__ bias,
                           float* __restrict__ Cf, u16* __restrict__ Cb,
                           int M, int N, int ldc) {
  int lane = threadIdx.x;
  int wid  = blockIdx.x * blockDim.y + threadIdx.y;
  int nstrips = M >> 6;
  int m0 = (wid % nstrips) << 6;
  int n0 = (wid / nstrips) << 4;
  int ntN = N >> 4;

  v8f c0 = {}, c1 = {}, c2 = {}, c3 = {};
  accum4(A0, lda0, B0, K0, ntN, m0, n0, lane, c0, c1, c2, c3);
  if (A1) accum4(A1, lda1, B1, K1, ntN, m0, n0, lane, c0, c1, c2, c3);
  if (A2) accum4(A2, lda2, B2, K2, ntN, m0, n0, lane, c0, c1, c2, c3);

  int n = n0 + (lane & 15);
  float bv = bias ? bias[n] : 0.0f;
  v8f accs[4] = {c0, c1, c2, c3};
  if (Cb) {
#pragma unroll
    for (int i = 0; i < 4; ++i) {
      int mb = m0 + (i << 4) + ((lane & 16) ? 8 : 0);
#pragma unroll
      for (int r = 0; r < 8; ++r)
        Cb[(long)(mb + r) * ldc + n] = f2bf(accs[i][r] + bv);
    }
  } else {
#pragma unroll
    for (int i = 0; i < 4; ++i) {
      int mb = m0 + (i << 4) + ((lane & 16) ? 8 : 0);
#pragma unroll
      for (int r = 0; r < 8; ++r)
        Cf[(long)(mb + r) * ldc + n] = accs[i][r] + bv;
    }
  }
}

// ---------------- B-matrix packing into WMMA fragment order -----------------
// packed[idx]: idx = tile*512 + lane*16 + e, tile = (k0/32)*ntN + (n0/16),
// lane<16: (k=k0+e, n=n0+lane) ; lane>=16: (k=k0+16+e, n=n0+lane-16)
__global__ void pack_b_nk(const float* __restrict__ W, u16* __restrict__ P,
                          int N, int K) {  // W is (N,K) row-major; B = W^T
  long idx = (long)blockIdx.x * blockDim.x + threadIdx.x;
  long tile = idx >> 9;
  int within = (int)(idx & 511);
  int lane = within >> 4, e = within & 15;
  int ntN = N >> 4;
  int kt = (int)(tile / ntN), nt = (int)(tile % ntN);
  int n = (nt << 4) + (lane & 15);
  int k = (kt << 5) + ((lane & 16) ? 16 : 0) + e;
  P[idx] = f2bf(W[(long)n * K + k]);
}
__global__ void pack_b_kn(const float* __restrict__ W, u16* __restrict__ P,
                          int N, int K) {  // W is (K,N) row-major; B = W
  long idx = (long)blockIdx.x * blockDim.x + threadIdx.x;
  long tile = idx >> 9;
  int within = (int)(idx & 511);
  int lane = within >> 4, e = within & 15;
  int ntN = N >> 4;
  int kt = (int)(tile / ntN), nt = (int)(tile % ntN);
  int n = (nt << 4) + (lane & 15);
  int k = (kt << 5) + ((lane & 16) ? 16 : 0) + e;
  P[idx] = f2bf(W[(long)k * N + n]);
}

// ---------------- elementwise / reduction kernels ---------------------------
__global__ void cvt_f2bf(const float* __restrict__ in, u16* __restrict__ out, long n) {
  long i = (long)blockIdx.x * blockDim.x + threadIdx.x;
  long stride = (long)gridDim.x * blockDim.x;
  for (; i < n; i += stride) out[i] = f2bf(in[i]);
}

__global__ void embed_kernel(const int* __restrict__ tok,
                             const float* __restrict__ table,
                             u16* __restrict__ out) {
  long idx = (long)blockIdx.x * blockDim.x + threadIdx.x; // B*L*EMB
  int e = (int)(idx & (EMB_ - 1));
  long bl = idx >> 9;
  out[idx] = f2bf(table[(long)tok[bl] * EMB_ + e]);
}

__global__ void addvec_kernel(const float* a, const float* b, float* o, int n) {
  int i = blockIdx.x * blockDim.x + threadIdx.x;
  if (i < n) o[i] = a[i] + b[i];
}

__global__ void zero_f32(float* p, long n) {
  long i = (long)blockIdx.x * blockDim.x + threadIdx.x;
  if (i < n) p[i] = 0.0f;
}
__global__ void zero_b16(u16* p, long n) {
  long i = (long)blockIdx.x * blockDim.x + threadIdx.x;
  if (i < n) p[i] = 0;
}

// score[b,t] = sum_a gvec[a] * tanh(enc_proj[b,t,a] + dproj[b,a])
__global__ void score_kernel(const u16* __restrict__ ep,
                             const float* __restrict__ dproj,
                             const float* __restrict__ gvec,
                             float* __restrict__ score) {
  __shared__ float sm[256];
  long bt = blockIdx.x;
  int b = (int)(bt >> 9);  // T=512
  const u16* row = ep + bt * 1024;
  const float* dp = dproj + (long)b * 1024;
  float s = 0.0f;
  for (int a = threadIdx.x; a < 1024; a += 256)
    s += gvec[a] * tanhf(bf2f(row[a]) + dp[a]);
  sm[threadIdx.x] = s;
  __syncthreads();
  for (int st = 128; st > 0; st >>= 1) {
    if (threadIdx.x < st) sm[threadIdx.x] += sm[threadIdx.x + st];
    __syncthreads();
  }
  if (threadIdx.x == 0) score[bt] = sm[0];
}

// masked softmax over T; writes scratch att_w and the (B,L,T) output slice
__global__ void softmax_kernel(const float* __restrict__ score,
                               const int* __restrict__ hlens,
                               float* __restrict__ attw,
                               float* __restrict__ att_out, int step) {
  __shared__ float sm[T_];
  int b = blockIdx.x, t = threadIdx.x;
  int hl = hlens[b];
  float s = (t < hl) ? score[b * T_ + t] : -1.0e10f;
  sm[t] = s;
  __syncthreads();
  for (int st = 256; st > 0; st >>= 1) {
    if (t < st) sm[t] = fmaxf(sm[t], sm[t + st]);
    __syncthreads();
  }
  float mx = sm[0];
  __syncthreads();
  float e = __expf(s - mx);
  sm[t] = e;
  __syncthreads();
  for (int st = 256; st > 0; st >>= 1) {
    if (t < st) sm[t] += sm[t + st];
    __syncthreads();
  }
  float w = e / sm[0];
  attw[b * T_ + t] = w;
  att_out[((long)b * L_ + step) * T_ + t] = w;
}

// att_c[b,d] = sum_t attw[b,t] * hs[b,t,d]
__global__ void attc_kernel(const float* __restrict__ attw,
                            const u16* __restrict__ hs,
                            float* __restrict__ attc,
                            u16* __restrict__ attc_bf) {
  int b = blockIdx.x, d = threadIdx.x;  // 512 threads: handles d and d+512
  const float* w = attw + b * T_;
  const u16* base = hs + (long)b * T_ * 1024;
  float a0 = 0.0f, a1 = 0.0f;
  for (int t = 0; t < T_; ++t) {
    float wt = w[t];
    const u16* r = base + (long)t * 1024;
    a0 += wt * bf2f(r[d]);
    a1 += wt * bf2f(r[d + 512]);
  }
  attc[b * 1024 + d] = a0;
  attc[b * 1024 + d + 512] = a1;
  attc_bf[b * 1024 + d] = f2bf(a0);
  attc_bf[b * 1024 + d + 512] = f2bf(a1);
}

// LSTM gate nonlinearity; gate order i,f,g,o in chunks of DUNITS
__global__ void lstm_act(const float* __restrict__ gates,
                         float* __restrict__ h, float* __restrict__ c,
                         u16* __restrict__ h_bf, u16* __restrict__ dec_bf,
                         int step) {
  int idx = blockIdx.x * blockDim.x + threadIdx.x;  // B*DUNITS = 65536
  int b = idx >> 10, d = idx & 1023;
  const float* g = gates + (long)b * NG_;
  float gi = sigm(g[d]);
  float gf = sigm(g[1024 + d]);
  float gg = tanhf(g[2048 + d]);
  float go = sigm(g[3072 + d]);
  float cn = gf * c[idx] + gi * gg;
  float hn = go * tanhf(cn);
  c[idx] = cn;
  h[idx] = hn;
  h_bf[idx] = f2bf(hn);
  if (dec_bf) dec_bf[((long)b * L_ + step) * DU_ + d] = f2bf(hn);
}

// ---------------------------------------------------------------------------
extern "C" void kernel_launch(void* const* d_in, const int* in_sizes, int n_in,
                              void* d_out, int out_size, void* d_ws, size_t ws_size,
                              hipStream_t stream) {
  const float* hs_pad = (const float*)d_in[0];
  const int*   hlens  = (const int*)d_in[1];
  const int*   tokens = (const int*)d_in[2];
  const float* emb_tb = (const float*)d_in[3];
  const float* w_enc  = (const float*)d_in[4];
  const float* b_enc  = (const float*)d_in[5];
  const float* w_dec  = (const float*)d_in[6];
  const float* gvec   = (const float*)d_in[7];
  const float* W_ih0  = (const float*)d_in[8];
  const float* W_hh0  = (const float*)d_in[9];
  const float* b_ih0  = (const float*)d_in[10];
  const float* b_hh0  = (const float*)d_in[11];
  const float* W_ih1  = (const float*)d_in[12];
  const float* W_hh1  = (const float*)d_in[13];
  const float* b_ih1  = (const float*)d_in[14];
  const float* b_hh1  = (const float*)d_in[15];
  const float* w_out  = (const float*)d_in[16];
  const float* b_out  = (const float*)d_in[17];

  float* logits_out = (float*)d_out;                         // (B*L, EMB)
  float* attw_out   = (float*)d_out + (long)B_ * L_ * EMB_;  // (B, L, T)

  // ---- carve workspace (bf16 keeps per-step working set ~L2-resident) -----
  size_t off = 0;
  char* base = (char*)d_ws;
  auto carve = [&](size_t bytes) -> char* {
    char* p = base + off;
    off += (bytes + 255) & ~(size_t)255;
    return p;
  };
  const long HS_N = (long)B_ * T_ * 1024;
  u16* hs_bf   = (u16*)carve(HS_N * 2);
  u16* ep_bf   = (u16*)carve(HS_N * 2);
  u16* wenc_pk = (u16*)carve(1024L * 1024 * 2);
  u16* wdec_pk = (u16*)carve(1024L * 1024 * 2);
  u16* wout_pk = (u16*)carve(1024L * 512 * 2);
  u16* wih0_pk = (u16*)carve(1536L * NG_ * 2);
  u16* whh0_pk = (u16*)carve(1024L * NG_ * 2);
  u16* wih1_pk = (u16*)carve(1024L * NG_ * 2);
  u16* whh1_pk = (u16*)carve(1024L * NG_ * 2);
  u16* emb_bf  = (u16*)carve((long)B_ * L_ * EMB_ * 2);
  u16* dec_bf  = (u16*)carve((long)B_ * L_ * DU_ * 2);
  float* hc    = (float*)carve(4L * B_ * DU_ * 4);  // h0,c0,h1,c1
  float* h0 = hc, *c0 = hc + 65536, *h1 = hc + 2 * 65536, *c1 = hc + 3 * 65536;
  u16* hbf     = (u16*)carve(2L * B_ * DU_ * 2);
  u16* h0_bf = hbf, *h1_bf = hbf + 65536;
  float* dproj = (float*)carve((long)B_ * 1024 * 4);
  float* score = (float*)carve((long)B_ * T_ * 4);
  float* attw  = (float*)carve((long)B_ * T_ * 4);
  float* attc  = (float*)carve((long)B_ * 1024 * 4);
  u16* attc_bf = (u16*)carve((long)B_ * 1024 * 2);
  float* gates = (float*)carve((long)B_ * NG_ * 4);
  float* bsum0 = (float*)carve(NG_ * 4);
  float* bsum1 = (float*)carve(NG_ * 4);

  dim3 gblk(32, 4);  // 4 waves per block, one 64x16 strip per wave

  // ---- one-time prep: conversions / packing / embedding / bias sums -------
  cvt_f2bf<<<4096, 256, 0, stream>>>(hs_pad, hs_bf, HS_N);
  pack_b_kn<<<(1024 * 1024) / 256, 256, 0, stream>>>(w_enc, wenc_pk, 1024, 1024);
  pack_b_kn<<<(1024 * 1024) / 256, 256, 0, stream>>>(w_dec, wdec_pk, 1024, 1024);
  pack_b_kn<<<(1024 * 512) / 256, 256, 0, stream>>>(w_out, wout_pk, 512, 1024);
  pack_b_nk<<<(NG_ * 1536) / 256, 256, 0, stream>>>(W_ih0, wih0_pk, NG_, 1536);
  pack_b_nk<<<(NG_ * 1024) / 256, 256, 0, stream>>>(W_hh0, whh0_pk, NG_, 1024);
  pack_b_nk<<<(NG_ * 1024) / 256, 256, 0, stream>>>(W_ih1, wih1_pk, NG_, 1024);
  pack_b_nk<<<(NG_ * 1024) / 256, 256, 0, stream>>>(W_hh1, whh1_pk, NG_, 1024);
  embed_kernel<<<(B_ * L_ * EMB_) / 256, 256, 0, stream>>>(tokens, emb_tb, emb_bf);
  addvec_kernel<<<16, 256, 0, stream>>>(b_ih0, b_hh0, bsum0, NG_);
  addvec_kernel<<<16, 256, 0, stream>>>(b_ih1, b_hh1, bsum1, NG_);
  zero_f32<<<1024, 256, 0, stream>>>(hc, 4L * B_ * DU_);
  zero_b16<<<512, 256, 0, stream>>>(hbf, 2L * B_ * DU_);

  // ---- enc_proj = hs_pad @ w_enc + b_enc  (bf16 out, stays L2 resident) ----
  // waves = (32768/64)*(1024/16) = 32768 -> 8192 blocks
  wmma_gemm3<<<8192, gblk, 0, stream>>>(
      hs_bf, 1024, wenc_pk, 1024,
      nullptr, 0, nullptr, 0, nullptr, 0, nullptr, 0,
      b_enc, nullptr, ep_bf, B_ * T_, 1024, 1024);

  // wih0 packed K-slabs: first 512 K-rows (emb part) then 1024 K-rows (att part)
  u16* wih0_pk_att = wih0_pk + (512L / 32) * (NG_ / 16) * 512;

  // ---- sequential decode over L steps ----
  for (int t = 0; t < L_; ++t) {
    // dproj = h0 @ w_dec : waves = 1*(1024/16)=64 -> 16 blocks
    wmma_gemm3<<<16, gblk, 0, stream>>>(
        h0_bf, 1024, wdec_pk, 1024,
        nullptr, 0, nullptr, 0, nullptr, 0, nullptr, 0,
        nullptr, dproj, nullptr, B_, 1024, 1024);
    // additive attention
    score_kernel<<<B_ * T_, 256, 0, stream>>>(ep_bf, dproj, gvec, score);
    softmax_kernel<<<B_, T_, 0, stream>>>(score, hlens, attw, attw_out, t);
    attc_kernel<<<B_, 512, 0, stream>>>(attw, hs_bf, attc, attc_bf);
    // layer-0 gates: [emb_t | att_c] @ W_ih0^T + h0 @ W_hh0^T + bias
    // waves = 1*(4096/16)=256 -> 64 blocks
    wmma_gemm3<<<64, gblk, 0, stream>>>(
        emb_bf + (long)t * EMB_, (long)L_ * EMB_, wih0_pk, EMB_,
        attc_bf, 1024, wih0_pk_att, 1024,
        h0_bf, 1024, whh0_pk, 1024,
        bsum0, gates, nullptr, B_, NG_, NG_);
    lstm_act<<<256, 256, 0, stream>>>(gates, h0, c0, h0_bf, nullptr, t);
    // layer-1 gates: h0_new @ W_ih1^T + h1 @ W_hh1^T + bias
    wmma_gemm3<<<64, gblk, 0, stream>>>(
        h0_bf, 1024, wih1_pk, 1024,
        h1_bf, 1024, whh1_pk, 1024,
        nullptr, 0, nullptr, 0,
        bsum1, gates, nullptr, B_, NG_, NG_);
    lstm_act<<<256, 256, 0, stream>>>(gates, h1, c1, h1_bf, dec_bf, t);
  }

  // ---- logits = dec @ w_out + b_out ----
  // waves = (8192/64)*(512/16) = 4096 -> 1024 blocks
  wmma_gemm3<<<1024, gblk, 0, stream>>>(
      dec_bf, 1024, wout_pk, 1024,
      nullptr, 0, nullptr, 0, nullptr, 0, nullptr, 0,
      b_out, logits_out, nullptr, B_ * L_, EMB_, EMB_);
}
